// PCFG_51625506897941
// MI455X (gfx1250) — matrix operand order repaired and checked by reading, not compiled, not run on hardware
//
#include <hip/hip_runtime.h>
#include <hip/hip_bf16.h>

typedef __attribute__((ext_vector_type(16))) _Float16 v16h;
typedef __attribute__((ext_vector_type(8)))  _Float16 v8h;
typedef __attribute__((ext_vector_type(8)))  float    v8f;

#define B_   8
#define N_   28
#define NT_  32
#define T_   64
#define ST_  96
#define RULE_K 9216            // ST_*ST_
#define NEGV (-1000000000.0f)

union Frag  { v16h v; v8h p[2]; _Float16 h[16]; };
union HFrag { v8h p; _Float16 h[8]; };
union Acc   { v8f v; float4 q[2]; };

// Permuted lr flattening = WMMA D-register order of GEMM1's 36 tiles:
//   K' = (li*6 + ri)*256 + (g*16 + lan)*8 + j   with l = li*16 + g*8 + j, r = ri*16 + lan
// Both numer and erule use K'; the lr contraction is invariant under this bijection.
__device__ __forceinline__ int kperm(int l, int r) {
    int li = l >> 4, lrem = l & 15;
    int g = lrem >> 3, j = lrem & 7;
    int ri = r >> 4, lan = r & 15;
    return ((li * 6 + ri) * 32 + (g * 16 + lan)) * 8 + j;
}

// ---------------- init beta: NEG everywhere, unary at width 0 ----------------
__global__ void pcfg_init(float* __restrict__ beta, const float* __restrict__ unary) {
    int idx = blockIdx.x * blockDim.x + threadIdx.x;
    const int total = B_ * N_ * N_ * ST_;
    if (idx >= total) return;
    int t = idx % ST_;
    int s = (idx / ST_) % N_;
    int w = (idx / (ST_ * N_)) % N_;
    int b = idx / (ST_ * N_ * N_);
    float v = NEGV;
    if (w == 0 && t >= NT_) v = unary[(b * N_ + s) * T_ + (t - NT_)];
    beta[idx] = v;
}

// ---------------- mrule + erule(f16, K'-permuted): one block per (b,a) ----------------
__global__ void pcfg_prep(const float* __restrict__ rule, float* __restrict__ mrule,
                          _Float16* __restrict__ erule) {
    int ba = blockIdx.x;                       // b*NT_ + a
    const float* rp = rule + (size_t)ba * RULE_K;
    __shared__ float red[256];
    float m = NEGV;
    for (int i = threadIdx.x; i < RULE_K; i += 256) m = fmaxf(m, rp[i]);
    red[threadIdx.x] = m;
    __syncthreads();
    for (int off = 128; off > 0; off >>= 1) {
        if (threadIdx.x < off) red[threadIdx.x] = fmaxf(red[threadIdx.x], red[threadIdx.x + off]);
        __syncthreads();
    }
    m = red[0];
    if (threadIdx.x == 0) mrule[ba] = m;
    _Float16* e = erule + (size_t)ba * RULE_K;
    for (int i = threadIdx.x; i < RULE_K; i += 256) {
        int l = i / ST_, r = i % ST_;
        e[kperm(l, r)] = (_Float16)__expf(rp[i] - m);   // one-time scatter into K' order
    }
}

// ---------------- GEMM1: numer[s] = el^T (96xK) @ er (Kx96), K=w padded to 32 ----------------
// one workgroup (8 waves) per (b,s); 36 WMMA tiles of the 96x96 result.
// A stored [l][k], B stored TRANSPOSED [r][k]: fragments = 2x ds_load_b128 each.
// D written in register order (K' layout) = 1x global_store_b128 per lane per tile.
__global__ void pcfg_numer(const float* __restrict__ beta, _Float16* __restrict__ numer,
                           float* __restrict__ Mspan, int w) {
    int S = N_ - w;
    int b = blockIdx.x / S;
    int s = blockIdx.x % S;
    __shared__ float    Lrow[32][ST_];
    __shared__ float    Rrow[32][ST_];
    __shared__ _Float16 Alds[ST_][32];      // A[l][k]   (A matrix 96 x 32, M-major)
    __shared__ _Float16 Bldst[ST_][32];     // B^T[r][k] (B matrix 32 x 96, stored N-major)
    __shared__ float    mlds[32], mrds[32];
    __shared__ float    Msh;

    int tid = threadIdx.x;
    // stage left/right beta rows for k = 0..w-1
    for (int i = tid; i < w * ST_; i += 256) {
        int k = i / ST_, c = i % ST_;
        Lrow[k][c] = beta[(((size_t)b * N_ + k) * N_ + s) * ST_ + c];
        Rrow[k][c] = beta[(((size_t)b * N_ + (w - 1 - k)) * N_ + (s + k + 1)) * ST_ + c];
    }
    __syncthreads();
    if (tid < w) {
        float ml = NEGV, mr = NEGV;
        for (int c = 0; c < ST_; ++c) {
            ml = fmaxf(ml, Lrow[tid][c]);
            mr = fmaxf(mr, Rrow[tid][c]);
        }
        mlds[tid] = ml; mrds[tid] = mr;
    }
    __syncthreads();
    if (tid == 0) {
        float M = NEGV;
        for (int k = 0; k < w; ++k) M = fmaxf(M, mlds[k] + mrds[k]);
        Msh = M;
        Mspan[b * N_ + s] = M;
    }
    __syncthreads();
    float M = Msh;
    // el = exp(left - ml);  er = exp(right - mr + (ml + mr - M)) = exp(right + ml - M)
    for (int i = tid; i < 32 * ST_; i += 256) {
        int k = i / ST_, c = i % ST_;
        _Float16 av = (_Float16)0.f, bv = (_Float16)0.f;
        if (k < w) {
            av = (_Float16)__expf(Lrow[k][c] - mlds[k]);
            bv = (_Float16)__expf(Rrow[k][c] + mlds[k] - M);
        }
        Alds[c][k]  = av;
        Bldst[c][k] = bv;
    }
    __syncthreads();

    int wave = tid >> 5, lane = tid & 31;
    int g = lane >> 4, lan = lane & 15;
    _Float16* out = numer + ((size_t)b * N_ + s) * RULE_K;
    for (int t = wave; t < 36; t += 8) {
        int li = t / 6, ri = t % 6;
        Frag af, bf;
        // A lane fragment: K = g*8 + {0..7} then 16 + g*8 + {0..7}  (contiguous)
        af.p[0] = *(const v8h*)&Alds[li * 16 + lan][g * 8];
        af.p[1] = *(const v8h*)&Alds[li * 16 + lan][16 + g * 8];
        // B lane fragment: K = g*8 + {0..7} then 16 + g*8 + {0..7} at fixed N = ri*16+lan
        bf.p[0] = *(const v8h*)&Bldst[ri * 16 + lan][g * 8];
        bf.p[1] = *(const v8h*)&Bldst[ri * 16 + lan][16 + g * 8];
        v8f c = {};
        c = __builtin_amdgcn_wmma_f32_16x16x32_f16(false, af.v, false, bf.v,
                                                   (short)0, c, false, false);
        // store in register order: K' = t*256 + lane*8 + j  -> single b128 store
        HFrag d;
#pragma unroll
        for (int j = 0; j < 8; ++j) d.h[j] = (_Float16)c[j];
        *(v8h*)(out + t * 256 + lane * 8) = d.p;
    }
}

// ---------------- GEMM2: total(SxNT) = numer(SxK') @ erule^T(K'xNT), K'=9216 ----------------
// one workgroup per b; 8 waves = 4 C-tiles x 2 K-halves; finalize log + write beta.
// K' permutation is transparent here; all fragments are contiguous global_load_b128.
__global__ void pcfg_total(const _Float16* __restrict__ numer, const _Float16* __restrict__ erule,
                           const float* __restrict__ mrule, const float* __restrict__ Mspan,
                           float* __restrict__ beta, int w) {
    int S = N_ - w;
    int b = blockIdx.x;
    int tid = threadIdx.x;
    int wave = tid >> 5, lane = tid & 31;
    int g = lane >> 4, lan = lane & 15;
    int ct = wave & 3, kh = wave >> 2;
    int mi = ct >> 1, ni = ct & 1;

    int srow = mi * 16 + lan;                    // span row (M)
    int acol = ni * 16 + lan;                    // nonterminal column (N)
    const _Float16* Abase = numer + ((size_t)b * N_ + srow) * RULE_K;
    const _Float16* Bbase = erule + ((size_t)b * NT_ + acol) * RULE_K;
    bool arow_ok = (srow < S);

    v8f c = {};
    const v8h zero8 = {};
    int k0 = kh * (RULE_K / 2);
#pragma unroll 4
    for (int kofs = 0; kofs < RULE_K / 2; kofs += 32) {
        int kk = k0 + kofs;
        __builtin_prefetch(Bbase + kk + 128, 0, 1);   // global_prefetch_b8: warm WGP$
        __builtin_prefetch(Abase + kk + 128, 0, 1);
        Frag af, bf;
        af.p[0] = arow_ok ? *(const v8h*)(Abase + kk + g * 8)      : zero8;
        af.p[1] = arow_ok ? *(const v8h*)(Abase + kk + 16 + g * 8) : zero8;
        bf.p[0] = *(const v8h*)(Bbase + kk + g * 8);
        bf.p[1] = *(const v8h*)(Bbase + kk + 16 + g * 8);
        c = __builtin_amdgcn_wmma_f32_16x16x32_f16(false, af.v, false, bf.v,
                                                   (short)0, c, false, false);
    }

    __shared__ float Cred[8][256];
    Acc cu; cu.v = c;
    ((float4*)&Cred[wave][lane * 8])[0] = cu.q[0];
    ((float4*)&Cred[wave][lane * 8])[1] = cu.q[1];
    __syncthreads();

    // combine the two K-halves, finalize score, write beta[:, w, s, :NT]
    for (int i = tid; i < 32 * NT_; i += 256) {
        int s = i >> 5, a = i & 31;
        if (s >= S) continue;
        int mi2 = s >> 4, mloc = s & 15;
        int ni2 = a >> 4, nloc = a & 15;
        int ct2 = mi2 * 2 + ni2;
        int lane2 = (mloc < 8) ? nloc : (16 + nloc);
        int j2    = (mloc < 8) ? mloc : (mloc - 8);
        float tot = Cred[ct2][lane2 * 8 + j2] + Cred[ct2 + 4][lane2 * 8 + j2];
        float score = __logf(tot) + Mspan[b * N_ + s] + mrule[b * NT_ + a];
        beta[(((size_t)b * N_ + w) * N_ + s) * ST_ + a] = score;
    }
}

// ---------------- final logsumexp: one wave32 per batch element ----------------
__global__ void pcfg_final(const float* __restrict__ beta, const float* __restrict__ root,
                           float* __restrict__ out) {
    int tid = threadIdx.x;
    int b = tid >> 5, a = tid & 31;
    float v = beta[(((size_t)b * N_ + (N_ - 1)) * N_ + 0) * ST_ + a] + root[b * NT_ + a];
    float m = v;
    for (int off = 16; off > 0; off >>= 1) m = fmaxf(m, __shfl_xor(m, off, 32));
    float e = __expf(v - m);
    for (int off = 16; off > 0; off >>= 1) e += __shfl_xor(e, off, 32);
    if (a == 0) out[b] = m + __logf(e);
}

extern "C" void kernel_launch(void* const* d_in, const int* in_sizes, int n_in,
                              void* d_out, int out_size, void* d_ws, size_t ws_size,
                              hipStream_t stream) {
    (void)in_sizes; (void)n_in; (void)out_size; (void)ws_size;
    const float* unary = (const float*)d_in[0];
    const float* rule  = (const float*)d_in[1];
    const float* root  = (const float*)d_in[2];
    float* out = (float*)d_out;

    char* ws = (char*)d_ws;
    size_t off = 0;
    float* beta = (float*)(ws + off);      off += (size_t)B_ * N_ * N_ * ST_ * sizeof(float);   // 2.4 MB
    float* mrule = (float*)(ws + off);     off += (size_t)B_ * NT_ * sizeof(float);             // 1 KB
    float* Mspan = (float*)(ws + off);     off += 1024;                                         // B*N padded
    _Float16* erule = (_Float16*)(ws + off); off += (size_t)B_ * NT_ * RULE_K * sizeof(_Float16); // 4.7 MB
    _Float16* numer = (_Float16*)(ws + off); off += (size_t)B_ * N_ * RULE_K * sizeof(_Float16);  // 4.1 MB

    pcfg_init<<<(B_ * N_ * N_ * ST_ + 255) / 256, 256, 0, stream>>>(beta, unary);
    pcfg_prep<<<B_ * NT_, 256, 0, stream>>>(rule, mrule, erule);
    for (int w = 1; w < N_; ++w) {
        int S = N_ - w;
        pcfg_numer<<<B_ * S, 256, 0, stream>>>(beta, numer, Mspan, w);
        pcfg_total<<<B_, 256, 0, stream>>>(numer, erule, mrule, Mspan, beta, w);
    }
    pcfg_final<<<1, 256, 0, stream>>>(beta, root, out);
}
